// SpeciesTreeGNNv2_22840636080297
// MI455X (gfx1250) — matrix operand order repaired and checked by reading, not compiled, not run on hardware
//
#include <hip/hip_runtime.h>

typedef _Float16 f16;
typedef _Float16 v16h __attribute__((ext_vector_type(16)));
typedef _Float16 v8h  __attribute__((ext_vector_type(8)));
typedef float    v8f  __attribute__((ext_vector_type(8)));

#define DEPTH 15
#define NNODE 65535      // 2^16 - 1
#define NINT  32767      // internal nodes
#define EPAIR 65534      // parent-child pairs
#define E2COL 131068     // edge_index columns (2*EPAIR)
#define MPAD  65536      // padded row count for activations
#define HID   128
#define HEADS 4

// ---------------------------------------------------------------------------
// Tree upsweep: avg over leaf descendants == recursive halving on perfect tree
// ---------------------------------------------------------------------------
__global__ void copy_nf_kernel(const float* __restrict__ nf, float* __restrict__ x13, int total) {
    int i = blockIdx.x * blockDim.x + threadIdx.x;
    if (i < total) x13[i] = nf[i];
}

__global__ void upsweep_kernel(float* __restrict__ x13, int s, int count) {
    int idx = blockIdx.x * blockDim.x + threadIdx.x;
    if (idx >= count * 13) return;
    int i = s + idx / 13;
    int f = idx % 13;
    float l = x13[(size_t)(2 * i + 1) * 13 + f];
    float r = x13[(size_t)(2 * i + 2) * 13 + f];
    x13[(size_t)i * 13 + f] = 0.5f * (l + r);
}

// ---------------------------------------------------------------------------
// Node MLP layer 1: (N,13) @ (13,128) + b, ReLU -> f16 (pad rows written 0)
// ---------------------------------------------------------------------------
__global__ void node_mlp1_kernel(const float* __restrict__ x13, const float* __restrict__ W1,
                                 const float* __restrict__ b1, f16* __restrict__ h1, int N) {
    int idx = blockIdx.x * blockDim.x + threadIdx.x;
    if (idx >= MPAD * HID) return;
    int n = idx >> 7, o = idx & 127;
    float s = 0.f;
    if (n < N) {
        #pragma unroll
        for (int k = 0; k < 13; ++k) s += x13[(size_t)n * 13 + k] * W1[k * HID + o];
        s += b1[o];
        s = fmaxf(s, 0.f);
    }
    h1[idx] = (f16)s;
}

// ---------------------------------------------------------------------------
// Pack weights (K x Nout, row-major f32) into WMMA B-fragment order, f16.
// Flat layout: ((nt*nKt + kt)*32 + lane)*16 + j ; lane h=lane>>4, r=lane&15:
// element j -> K = kt*32 + 8*h + j (j<8), kt*32 + 16 + 8*h + (j-8) (j>=8), col = nt*16 + r. Zero-pad K >= Kreal.
// ---------------------------------------------------------------------------
__global__ void pack_b_kernel(const float* __restrict__ W, f16* __restrict__ Bp,
                              int Kreal, int Nout, int nKt) {
    int idx = blockIdx.x * blockDim.x + threadIdx.x;
    int total = (Nout / 16) * nKt * 512;
    if (idx >= total) return;
    int j = idx & 15;
    int lane = (idx >> 4) & 31;
    int t = idx >> 9;
    int kt = t % nKt;
    int nt = t / nKt;
    int h = lane >> 4, r = lane & 15;
    int k = kt * 32 + 8 * h + (j < 8 ? j : 8 + j);
    int col = nt * 16 + r;
    Bp[idx] = (k < Kreal) ? (f16)W[(size_t)k * Nout + col] : (f16)0.f;
}

// ---------------------------------------------------------------------------
// WMMA GEMM: C(MxNout,f32 opt) / Ch(f16 opt) = act(A(f16,MxKp) @ Bp + bias)
// One wave -> 16x64 strip: 4x v_wmma_f32_16x16x32_f16 per K-chunk, shared A.
// ---------------------------------------------------------------------------
__global__ void __launch_bounds__(256)
gemm_wmma_kernel(const f16* __restrict__ A, const f16* __restrict__ Bp,
                 const float* __restrict__ bias,
                 float* __restrict__ Cf, f16* __restrict__ Ch,
                 int M, int Kp, int Nout, int relu) {
    const int nKt = Kp >> 5;
    const int Ngrp = Nout >> 6;                 // groups of 64 cols
    const int Mtiles = (M + 15) >> 4;
    int w = blockIdx.x * (blockDim.x >> 5) + (threadIdx.x >> 5);
    if (w >= Mtiles * Ngrp) return;
    int mt = w / Ngrp;
    int ng = w - mt * Ngrp;
    int m0 = mt << 4;
    int n0 = ng << 6;

    int lane = threadIdx.x & 31;
    int h = lane >> 4, r = lane & 15;

    const f16* arow = A + (size_t)(m0 + r) * Kp;  // rows padded to MPAD: no guard
    const f16* bbase0 = Bp + (size_t)lane * 16;

    v8f z = {0.f, 0.f, 0.f, 0.f, 0.f, 0.f, 0.f, 0.f};
    v8f c0 = z, c1 = z, c2 = z, c3 = z;

    for (int kt = 0; kt < nKt; ++kt) {
        int k0 = kt << 5;
        v8h alo = *(const v8h*)(arow + k0 + 8 * h);
        v8h ahi = *(const v8h*)(arow + k0 + 16 + 8 * h);
        v16h a;
        #pragma unroll
        for (int j = 0; j < 8; ++j) { a[j] = alo[j]; a[j + 8] = ahi[j]; }

        const f16* bb = bbase0 + (size_t)kt * 512;
        size_t gstride = (size_t)nKt * 512;
        v16h b0 = *(const v16h*)(bb + (size_t)(ng * 4 + 0) * gstride);
        v16h b1v = *(const v16h*)(bb + (size_t)(ng * 4 + 1) * gstride);
        v16h b2v = *(const v16h*)(bb + (size_t)(ng * 4 + 2) * gstride);
        v16h b3v = *(const v16h*)(bb + (size_t)(ng * 4 + 3) * gstride);

        c0 = __builtin_amdgcn_wmma_f32_16x16x32_f16(false, a, false, b0,  (short)0, c0, false, false);
        c1 = __builtin_amdgcn_wmma_f32_16x16x32_f16(false, a, false, b1v, (short)0, c1, false, false);
        c2 = __builtin_amdgcn_wmma_f32_16x16x32_f16(false, a, false, b2v, (short)0, c2, false, false);
        c3 = __builtin_amdgcn_wmma_f32_16x16x32_f16(false, a, false, b3v, (short)0, c3, false, false);
    }

    v8f acc[4] = {c0, c1, c2, c3};
    #pragma unroll
    for (int g = 0; g < 4; ++g) {
        int col = n0 + g * 16 + r;
        float bv = bias ? bias[col] : 0.f;
        #pragma unroll
        for (int v = 0; v < 8; ++v) {
            int row = m0 + v + 8 * h;
            if (row < M) {
                float val = acc[g][v] + bv;
                if (relu) val = fmaxf(val, 0.f);
                if (Cf) Cf[(size_t)row * Nout + col] = val;
                if (Ch) Ch[(size_t)row * Nout + col] = (f16)val;
            }
        }
    }
}

// ---------------------------------------------------------------------------
// GAT attention scores: e_src/e_dst (N,4) from xp (N,4,128) f16
// ---------------------------------------------------------------------------
__global__ void gat_scores_kernel(const f16* __restrict__ xp, const float* __restrict__ asrc,
                                  const float* __restrict__ adst,
                                  float* __restrict__ es, float* __restrict__ ed, int N) {
    int idx = blockIdx.x * blockDim.x + threadIdx.x;
    if (idx >= N * HEADS) return;
    int n = idx >> 2, hh = idx & 3;
    const f16* xr = xp + (size_t)n * (HEADS * HID) + hh * HID;
    float s1 = 0.f, s2 = 0.f;
    #pragma unroll 4
    for (int d = 0; d < HID; ++d) {
        float xv = (float)xr[d];
        s1 += xv * asrc[hh * HID + d];
        s2 += xv * adst[hh * HID + d];
    }
    es[idx] = s1;
    ed[idx] = s2;
}

// ---------------------------------------------------------------------------
// Fused GAT attention (gather, max in-degree 4) + mean-heads + bias +
// residual + LayerNorm. One block (128 threads) per node.
// ---------------------------------------------------------------------------
__global__ void __launch_bounds__(128)
gat_attn_ln_kernel(const f16* __restrict__ xp, const float* __restrict__ es,
                   const float* __restrict__ ed, const float* __restrict__ gb,
                   const float* __restrict__ lng, const float* __restrict__ lnb,
                   float* __restrict__ x, f16* __restrict__ xh) {
    int i = blockIdx.x;
    int t = threadIdx.x;

    int nbr[4];
    int cnt = 0;
    nbr[cnt++] = i;                              // self loop
    if (i > 0) nbr[cnt++] = (i - 1) >> 1;        // parent -> i
    if (i < NINT) { nbr[cnt++] = 2 * i + 1; nbr[cnt++] = 2 * i + 2; }  // children -> i

    float out = 0.f;
    #pragma unroll
    for (int hh = 0; hh < HEADS; ++hh) {
        float edi = ed[i * HEADS + hh];
        float sc[4];
        float mx = -1e30f;
        for (int j = 0; j < cnt; ++j) {
            float s = es[nbr[j] * HEADS + hh] + edi;
            s = (s > 0.f) ? s : 0.2f * s;        // leaky_relu 0.2
            sc[j] = s;
            mx = fmaxf(mx, s);
        }
        float den = 0.f;
        for (int j = 0; j < cnt; ++j) { sc[j] = __expf(sc[j] - mx); den += sc[j]; }
        float inv = 1.f / den;
        float acc = 0.f;
        for (int j = 0; j < cnt; ++j)
            acc += sc[j] * (float)xp[(size_t)nbr[j] * (HEADS * HID) + hh * HID + t];
        out += acc * inv;
    }
    out *= (1.f / HEADS);
    out += gb[t] + x[(size_t)i * HID + t];       // bias + residual

    __shared__ float red[HID];
    red[t] = out;
    __syncthreads();
    for (int s2 = 64; s2 > 0; s2 >>= 1) { if (t < s2) red[t] += red[t + s2]; __syncthreads(); }
    float mean = red[0] * (1.f / HID);
    __syncthreads();
    float dv = out - mean;
    red[t] = dv * dv;
    __syncthreads();
    for (int s2 = 64; s2 > 0; s2 >>= 1) { if (t < s2) red[t] += red[t + s2]; __syncthreads(); }
    float var = red[0] * (1.f / HID);
    float xn = dv * rsqrtf(var + 1e-5f) * lng[t] + lnb[t];
    x[(size_t)i * HID + t] = xn;
    xh[(size_t)i * HID + t] = (f16)xn;
}

// ---------------------------------------------------------------------------
// Build edge-MLP input: [x[se] | x[de] | ef] as f16, K padded 260 -> 288
// ---------------------------------------------------------------------------
__global__ void build_ein_kernel(const float* __restrict__ x, const int* __restrict__ ei,
                                 const float* __restrict__ ef, f16* __restrict__ ein) {
    int idx = blockIdx.x * blockDim.x + threadIdx.x;
    if (idx >= MPAD * 288) return;
    int e = idx / 288;
    int k = idx - e * 288;
    f16 v = (f16)0.f;
    if (e < EPAIR) {
        if (k < 128) {
            int se = ei[2 * e];
            v = (f16)x[(size_t)se * HID + k];
        } else if (k < 256) {
            int de = ei[(size_t)E2COL + 2 * e];
            v = (f16)x[(size_t)de * HID + (k - 128)];
        } else if (k < 260) {
            v = (f16)ef[(size_t)e * 4 + (k - 256)];
        }
    }
    ein[idx] = v;
}

// ---------------------------------------------------------------------------
// Head layer 2: (E,128) @ (128,2) + b -> logits f32
// ---------------------------------------------------------------------------
__global__ void wh2_kernel(const f16* __restrict__ hh, const float* __restrict__ W2,
                           const float* __restrict__ b2, float* __restrict__ out) {
    int idx = blockIdx.x * blockDim.x + threadIdx.x;
    if (idx >= EPAIR * 2) return;
    int e = idx >> 1, o = idx & 1;
    float s = b2[o];
    #pragma unroll 4
    for (int k = 0; k < HID; ++k) s += (float)hh[(size_t)e * HID + k] * W2[k * 2 + o];
    out[idx] = s;
}

// ---------------------------------------------------------------------------
static inline size_t alloc_off(size_t& off, size_t bytes) {
    size_t cur = off;
    off = (off + bytes + 255) & ~(size_t)255;
    return cur;
}

extern "C" void kernel_launch(void* const* d_in, const int* in_sizes, int n_in,
                              void* d_out, int out_size, void* d_ws, size_t ws_size,
                              hipStream_t stream) {
    (void)in_sizes; (void)n_in; (void)out_size; (void)ws_size;
    const float* nf    = (const float*)d_in[0];
    const int*   ei    = (const int*)d_in[1];
    const float* ef    = (const float*)d_in[2];
    const float* np_W1 = (const float*)d_in[4];
    const float* np_b1 = (const float*)d_in[5];
    const float* np_W2 = (const float*)d_in[6];
    const float* np_b2 = (const float*)d_in[7];
    const float* gat_W = (const float*)d_in[8];
    const float* gat_as= (const float*)d_in[9];
    const float* gat_ad= (const float*)d_in[10];
    const float* gat_b = (const float*)d_in[11];
    const float* ln_g  = (const float*)d_in[12];
    const float* ln_b  = (const float*)d_in[13];
    const float* em_W1 = (const float*)d_in[14];
    const float* em_b1 = (const float*)d_in[15];
    const float* em_W2 = (const float*)d_in[16];
    const float* em_b2 = (const float*)d_in[17];
    const float* wh_W1 = (const float*)d_in[18];
    const float* wh_b1 = (const float*)d_in[19];
    const float* wh_W2 = (const float*)d_in[20];
    const float* wh_b2 = (const float*)d_in[21];

    char* ws = (char*)d_ws;
    size_t off = 0;
    float* x13  = (float*)(ws + alloc_off(off, (size_t)MPAD * 13 * 4));
    float* x    = (float*)(ws + alloc_off(off, (size_t)MPAD * HID * 4));
    f16*   xh   = (f16*)  (ws + alloc_off(off, (size_t)MPAD * HID * 2));
    f16*   xp   = (f16*)  (ws + alloc_off(off, (size_t)MPAD * 512 * 2));
    f16*   h1   = (f16*)  (ws + alloc_off(off, (size_t)MPAD * HID * 2));
    float* es   = (float*)(ws + alloc_off(off, (size_t)MPAD * 4 * 4));
    float* ed   = (float*)(ws + alloc_off(off, (size_t)MPAD * 4 * 4));
    f16*   ein  = (f16*)  (ws + alloc_off(off, (size_t)MPAD * 288 * 2));
    f16*   eh   = (f16*)  (ws + alloc_off(off, (size_t)MPAD * HID * 2));
    f16*   embh = (f16*)  (ws + alloc_off(off, (size_t)MPAD * HID * 2));
    f16*   whh  = (f16*)  (ws + alloc_off(off, (size_t)MPAD * HID * 2));
    f16*   Bnp2 = (f16*)  (ws + alloc_off(off, (size_t)8  * 4 * 512 * 2));
    f16*   Bgat0= (f16*)  (ws + alloc_off(off, (size_t)32 * 4 * 512 * 2));
    f16*   Bgat1= (f16*)  (ws + alloc_off(off, (size_t)32 * 4 * 512 * 2));
    f16*   Bgat2= (f16*)  (ws + alloc_off(off, (size_t)32 * 4 * 512 * 2));
    f16*   Bem1 = (f16*)  (ws + alloc_off(off, (size_t)8  * 9 * 512 * 2));
    f16*   Bem2 = (f16*)  (ws + alloc_off(off, (size_t)8  * 4 * 512 * 2));
    f16*   Bwh1 = (f16*)  (ws + alloc_off(off, (size_t)8  * 4 * 512 * 2));

    const int TB = 256;
    // --- pack all WMMA B operands (f32 -> f16 fragment order) ---
    pack_b_kernel<<<(8 * 4 * 512 + TB - 1) / TB, TB, 0, stream>>>(np_W2, Bnp2, 128, 128, 4);
    pack_b_kernel<<<(32 * 4 * 512 + TB - 1) / TB, TB, 0, stream>>>(gat_W + 0 * 128 * 512, Bgat0, 128, 512, 4);
    pack_b_kernel<<<(32 * 4 * 512 + TB - 1) / TB, TB, 0, stream>>>(gat_W + 1 * 128 * 512, Bgat1, 128, 512, 4);
    pack_b_kernel<<<(32 * 4 * 512 + TB - 1) / TB, TB, 0, stream>>>(gat_W + 2 * 128 * 512, Bgat2, 128, 512, 4);
    pack_b_kernel<<<(8 * 9 * 512 + TB - 1) / TB, TB, 0, stream>>>(em_W1, Bem1, 260, 128, 9);
    pack_b_kernel<<<(8 * 4 * 512 + TB - 1) / TB, TB, 0, stream>>>(em_W2, Bem2, 128, 128, 4);
    pack_b_kernel<<<(8 * 4 * 512 + TB - 1) / TB, TB, 0, stream>>>(wh_W1, Bwh1, 128, 128, 4);

    // --- tree propagation (bottom-up average upsweep) ---
    copy_nf_kernel<<<(NNODE * 13 + TB - 1) / TB, TB, 0, stream>>>(nf, x13, NNODE * 13);
    for (int d = DEPTH - 1; d >= 0; --d) {
        int s = (1 << d) - 1;
        int count = 1 << d;
        upsweep_kernel<<<(count * 13 + TB - 1) / TB, TB, 0, stream>>>(x13, s, count);
    }

    // --- node MLP: 13->128 (VALU) then 128->128 (WMMA) ---
    node_mlp1_kernel<<<((size_t)MPAD * HID + TB - 1) / TB, TB, 0, stream>>>(x13, np_W1, np_b1, h1, NNODE);
    {
        int waves = ((NNODE + 15) / 16) * (128 / 64);
        gemm_wmma_kernel<<<(waves + 7) / 8, TB, 0, stream>>>(h1, Bnp2, np_b2, x, xh, NNODE, 128, 128, 0);
    }

    // --- 3 GAT layers ---
    f16* Bg[3] = {Bgat0, Bgat1, Bgat2};
    for (int l = 0; l < 3; ++l) {
        int waves = ((NNODE + 15) / 16) * (512 / 64);
        gemm_wmma_kernel<<<(waves + 7) / 8, TB, 0, stream>>>(xh, Bg[l], nullptr, nullptr, xp, NNODE, 128, 512, 0);
        gat_scores_kernel<<<(NNODE * HEADS + TB - 1) / TB, TB, 0, stream>>>(
            xp, gat_as + (size_t)l * HEADS * HID, gat_ad + (size_t)l * HEADS * HID, es, ed, NNODE);
        gat_attn_ln_kernel<<<NNODE, HID, 0, stream>>>(
            xp, es, ed, gat_b + (size_t)l * HID, ln_g + (size_t)l * HID, ln_b + (size_t)l * HID, x, xh);
    }

    // --- edge MLP: build [x[se]|x[de]|ef] (K=260 pad 288), two WMMA GEMMs ---
    build_ein_kernel<<<((size_t)MPAD * 288 + TB - 1) / TB, TB, 0, stream>>>(x, ei, ef, ein);
    {
        int waves = ((EPAIR + 15) / 16) * (128 / 64);
        gemm_wmma_kernel<<<(waves + 7) / 8, TB, 0, stream>>>(ein, Bem1, em_b1, nullptr, eh, EPAIR, 288, 128, 1);
        float* emb_out = (float*)d_out + (size_t)EPAIR * 2;  // output 1: emb (E,128) f32
        gemm_wmma_kernel<<<(waves + 7) / 8, TB, 0, stream>>>(eh, Bem2, em_b2, emb_out, embh, EPAIR, 128, 128, 0);
        // --- head: 128->128 (WMMA, ReLU) then 128->2 (VALU) ---
        gemm_wmma_kernel<<<(waves + 7) / 8, TB, 0, stream>>>(embh, Bwh1, wh_b1, nullptr, whh, EPAIR, 128, 128, 1);
    }
    wh2_kernel<<<(EPAIR * 2 + TB - 1) / TB, TB, 0, stream>>>(whh, wh_W2, wh_b2, (float*)d_out);
}
